// WindowAttention3D_67190468379226
// MI455X (gfx1250) — matrix operand order repaired and checked by reading, not compiled, not run on hardware
//
#include <hip/hip_runtime.h>

typedef __attribute__((ext_vector_type(16))) _Float16 v16h;
typedef __attribute__((ext_vector_type(8)))  float    v8f;

#define SCALE 0.17677669529663687f

// workspace layout (bytes)
#define QSZ     (4ULL*4096*64*32*2)        // q/k/v each: [head][win][64][32] f16 = 64 MiB
#define OFF_Q   0ULL
#define OFF_K   (OFF_Q + QSZ)
#define OFF_V   (OFF_K + QSZ)
#define OFF_AT  (OFF_V + QSZ)              // attn out f16 [win][64][128]
#define ATSZ    (4096ULL*64*128*2)
#define OFF_WQ  (OFF_AT + ATSZ)            // Wqkv packed B-frags: [24][4][32][8] u32
#define WQSZ    (24ULL*4*32*8*4)
#define OFF_WO  (OFF_WQ + WQSZ)            // Wout packed B-frags: [8][4][32][8] u32
#define WOSZ    (8ULL*4*32*8*4)
#define OFF_B   (OFF_WO + WOSZ)            // bias f32 [64][64]

union Frag { v16h v; unsigned u[8]; };

__device__ inline unsigned pack_f16x2(float a, float b) {
    union { _Float16 h[2]; unsigned u; } t;
    t.h[0] = (_Float16)a; t.h[1] = (_Float16)b;
    return t.u;
}

__device__ inline v8f wmma_f16(v16h A, v16h B, v8f C) {
    return __builtin_amdgcn_wmma_f32_16x16x32_f16(false, A, false, B, (short)0, C, false, false);
}

// CDNA5 async global->LDS copy (per-lane b128), tracked by ASYNCcnt.
__device__ inline void async_ld_b128(void* lds, const void* g) {
    unsigned l = (unsigned)(unsigned long long)(__attribute__((address_space(3))) char*)lds;
    asm volatile("global_load_async_to_lds_b128 %0, %1, off"
                 :: "v"(l), "v"((unsigned long long)g) : "memory");
}
__device__ inline void wait_async0() {
    asm volatile("s_wait_asynccnt 0x0" ::: "memory");
}

// ---------------------------------------------------------------------------
// Kernel 0: pack weights into WMMA B-fragment order (f16) + build bias table.
// B-fragment order used here (16-bit, 32x16): lane l holds col n = l&15; vgpr v holds K pair
// k = 2v + 16*(l>=16).  Packed index: [(tile*4+kk)*32 + l]*8 + v   (u32 each)
// ---------------------------------------------------------------------------
__global__ void prep_kernel(const float* __restrict__ Wqkv,
                            const float* __restrict__ Wout,
                            const float* __restrict__ pos,
                            unsigned* __restrict__ wqf,
                            unsigned* __restrict__ wof,
                            float* __restrict__ bias) {
    int id = blockIdx.x * blockDim.x + threadIdx.x;
    if (id < 24576) {                    // Wqkv: 24 N-tiles x 4 K-steps
        int r = id & 7, l = (id >> 3) & 31, kk = (id >> 8) & 3, t = id >> 10;
        int k = kk * 32 + 2 * r + ((l >= 16) ? 16 : 0);
        int n = t * 16 + (l & 15);
        wqf[id] = pack_f16x2(Wqkv[k * 384 + n], Wqkv[(k + 1) * 384 + n]);
    } else if (id < 24576 + 8192) {      // Wout: 8 N-tiles x 4 K-steps
        int j = id - 24576;
        int r = j & 7, l = (j >> 3) & 31, kk = (j >> 8) & 3, t = j >> 10;
        int k = kk * 32 + 2 * r + ((l >= 16) ? 16 : 0);
        int n = t * 16 + (l & 15);
        wof[j] = pack_f16x2(Wout[k * 128 + n], Wout[(k + 1) * 128 + n]);
    } else if (id < 24576 + 8192 + 4096) {  // relative position bias [64][64]
        int j = id - 32768;
        int i = j >> 6, jj = j & 63;
        int iz = i >> 4, iy = (i >> 2) & 3, ix = i & 3;
        int jz = jj >> 4, jy = (jj >> 2) & 3, jx = jj & 3;
        int r0 = iz - jz + 3, r1 = iy - jy + 3, r2 = ix - jx + 3;
        bias[j] = pos[(r0 * 7 + r1) * 7 + r2];
    }
}

// ---------------------------------------------------------------------------
// Kernel 1: fused cyclic-shift + QKV projection.  One block per window.
// 8 waves; wave = (wm in 0..3 rows, half in 0..1 of 24 N-tiles).
// ---------------------------------------------------------------------------
__global__ __launch_bounds__(256)
void qkv_kernel(const float* __restrict__ x, const unsigned* __restrict__ wqf,
                _Float16* __restrict__ qo, _Float16* __restrict__ ko,
                _Float16* __restrict__ vo) {
    __shared__ _Float16 Xs[64][128];
    int w = blockIdx.x;
    int wh = w >> 8, ww = (w >> 4) & 15, wd = w & 15;
    int tid = threadIdx.x;

    __builtin_prefetch(wqf, 0, 2);      // L2-resident packed weights

    // gather shifted 64x128 tile -> LDS f16 (roll by -2 folded into index)
    for (int it = 0; it < 8; ++it) {
        int idx = tid + it * 256;            // 0..2047 float4 slots
        int r = idx >> 5, c4 = idx & 31;
        int iz = r >> 4, iy = (r >> 2) & 3, ix = r & 3;
        int z = (wh * 4 + iz + 2) & 63;
        int y = (ww * 4 + iy + 2) & 63;
        int xd = (wd * 4 + ix + 2) & 63;
        float4 f = *(const float4*)(x + ((((size_t)z * 64 + y) * 64 + xd) * 128) + c4 * 4);
        Xs[r][c4 * 4 + 0] = (_Float16)f.x; Xs[r][c4 * 4 + 1] = (_Float16)f.y;
        Xs[r][c4 * 4 + 2] = (_Float16)f.z; Xs[r][c4 * 4 + 3] = (_Float16)f.w;
    }
    __syncthreads();

    int wave = tid >> 5, l = tid & 31;
    int wm = wave >> 1, half = wave & 1;
    int lrow = l & 15, hi = (l >= 16) ? 1 : 0;
    int m = 16 * wm + lrow;

    // hoist A fragments for all 4 K-steps (contiguous-k A-layout, 16-bit 16x32)
    Frag a[4];
    for (int kk = 0; kk < 4; ++kk)
        for (int r = 0; r < 8; ++r) {
            int kb = kk * 32 + 2 * r + (hi ? 16 : 0);
            a[kk].u[r] = *(const unsigned*)&Xs[m][kb];
        }

    for (int t0 = 0; t0 < 12; ++t0) {
        int t = half * 12 + t0;
        v8f acc = {};
        for (int kk = 0; kk < 4; ++kk) {
            Frag b;
            const uint4* bp = (const uint4*)(wqf + ((size_t)(t * 4 + kk) * 32 + l) * 8);
            *(uint4*)&b.u[0] = bp[0];
            *(uint4*)&b.u[4] = bp[1];
            acc = wmma_f16(a[kk].v, b.v, acc);
        }
        int col = t * 16 + lrow;
        int which = col >> 7, within = col & 127;
        int head = within >> 5, hd = within & 31;
        _Float16* base = (which == 0) ? qo : ((which == 1) ? ko : vo);
        size_t o = (((size_t)head * 4096 + w) * 64) * 32 + hd;
        for (int r = 0; r < 8; ++r) {
            int mm = 16 * wm + r + (hi ? 8 : 0);
            base[o + (size_t)mm * 32] = (_Float16)acc[r];
        }
    }
}

// ---------------------------------------------------------------------------
// Kernel 2: windowed attention.  One block per (window, head), 4 waves.
// Q/K/V/bias staged with CDNA5 async global->LDS DMA (ASYNCcnt).
// ---------------------------------------------------------------------------
__global__ __launch_bounds__(128)
void attn_kernel(const _Float16* __restrict__ qws, const _Float16* __restrict__ kws,
                 const _Float16* __restrict__ vws, const float* __restrict__ bias,
                 _Float16* __restrict__ aout) {
    __shared__ _Float16 Qs[64][32];
    __shared__ _Float16 Ks[64][32];
    __shared__ _Float16 Vraw[64][32];    // row-major V staging
    __shared__ _Float16 Vsw[32][64];     // transposed V: Vsw[n][k] = V[k][n]
    __shared__ float    Sb[64][64];
    __shared__ _Float16 Pb[64][64];
    __shared__ float    Bs[64][64];

    int bid = blockIdx.x;
    int w = bid >> 2, h = bid & 3;
    int tid = threadIdx.x;
    size_t hb = (((size_t)h * 4096 + w) * 64) * 32;
    const _Float16* qp = qws + hb;
    const _Float16* kp = kws + hb;
    const _Float16* vp = vws + hb;

    // async DMA: 256 b128 per 4KB matrix, 1024 b128 for the 16KB bias table
    for (int it = 0; it < 2; ++it) {
        int idx = tid + it * 128;
        async_ld_b128(&((uint4*)Qs)[idx],   (const uint4*)qp + idx);
        async_ld_b128(&((uint4*)Ks)[idx],   (const uint4*)kp + idx);
        async_ld_b128(&((uint4*)Vraw)[idx], (const uint4*)vp + idx);
    }
    for (int it = 0; it < 8; ++it) {
        int idx = tid + it * 128;
        async_ld_b128(&((float4*)Bs)[idx], (const float4*)bias + idx);
    }
    wait_async0();
    __syncthreads();

    // transpose V once in LDS so PV B-fragments are contiguous dword reads
    for (int it = 0; it < 16; ++it) {
        int idx = tid + it * 128;
        Vsw[idx & 31][idx >> 5] = Vraw[idx >> 5][idx & 31];
    }

    int wave = tid >> 5, l = tid & 31;
    int lrow = l & 15, hi = (l >= 16) ? 1 : 0;
    int m0 = wave * 16;
    int wh = w >> 8, wwi = (w >> 4) & 15, wd = w & 15;
    bool mul = (wh == 15);
    bool mlr = (wwi == 15) && (wd == 15);
    bool mdp = (wd == 15);

    // S = Q K^T  (K-dim 32 -> single WMMA per 16x16 tile)
    Frag a;
    for (int r = 0; r < 8; ++r) {
        int kb = 2 * r + (hi ? 16 : 0);
        a.u[r] = *(const unsigned*)&Qs[m0 + lrow][kb];
    }
    for (int nt = 0; nt < 4; ++nt) {
        Frag b;
        for (int r = 0; r < 8; ++r) {
            int k = 2 * r + (hi ? 16 : 0);
            b.u[r] = *(const unsigned*)&Ks[nt * 16 + lrow][k];   // B[k][n] = K[n][k]
        }
        v8f acc = {};
        acc = wmma_f16(a.v, b.v, acc);
        int n = nt * 16 + lrow;
        for (int r = 0; r < 8; ++r) {
            int m = m0 + r + (hi ? 8 : 0);
            float s = acc[r] * SCALE + Bs[m][n];
            if (mul && ((m >= 32) != (n >= 32)))                       s = -1e30f;
            if (mlr && ((((m >> 2) & 3) >= 2) != (((n >> 2) & 3) >= 2))) s = -1e30f;
            if (mdp && (((m & 3) >= 2) != ((n & 3) >= 2)))             s = -1e30f;
            Sb[m][n] = s;
        }
    }
    __syncthreads();

    // softmax, one row per thread (64 rows)
    if (tid < 64) {
        float mx = -3.0e38f;
        for (int j = 0; j < 64; ++j) mx = fmaxf(mx, Sb[tid][j]);
        float sum = 0.f;
        for (int j = 0; j < 64; ++j) { float e = __expf(Sb[tid][j] - mx); Sb[tid][j] = e; sum += e; }
        float inv = 1.0f / sum;
        for (int j = 0; j < 64; ++j) Pb[tid][j] = (_Float16)(Sb[tid][j] * inv);
    }
    __syncthreads();

    // O = P V   (K-dim 64 -> two chained WMMAs per tile)
    Frag a0, a1;
    for (int r = 0; r < 8; ++r) {
        int kb = 2 * r + (hi ? 16 : 0);
        a0.u[r] = *(const unsigned*)&Pb[m0 + lrow][kb];
        a1.u[r] = *(const unsigned*)&Pb[m0 + lrow][kb + 32];
    }
    for (int nt = 0; nt < 2; ++nt) {
        Frag b0, b1;
        int nn = nt * 16 + lrow;
        for (int r = 0; r < 8; ++r) {
            int k = 2 * r + (hi ? 16 : 0);
            b0.u[r] = *(const unsigned*)&Vsw[nn][k];
            b1.u[r] = *(const unsigned*)&Vsw[nn][k + 32];
        }
        v8f acc = {};
        acc = wmma_f16(a0.v, b0.v, acc);
        acc = wmma_f16(a1.v, b1.v, acc);
        for (int r = 0; r < 8; ++r) {
            int m = m0 + r + (hi ? 8 : 0);
            aout[(((size_t)w * 64 + m) * 128) + h * 32 + nn] = (_Float16)acc[r];
        }
    }
}

// ---------------------------------------------------------------------------
// Kernel 3: output projection + bias + inverse cyclic shift scatter (f32 out).
// Activation tile async-staged to LDS.
// ---------------------------------------------------------------------------
__global__ __launch_bounds__(128)
void proj_kernel(const _Float16* __restrict__ at, const unsigned* __restrict__ wof,
                 const float* __restrict__ bout, float* __restrict__ out) {
    __shared__ _Float16 As[64][128];
    int w = blockIdx.x;
    int wh = w >> 8, ww = (w >> 4) & 15, wd = w & 15;
    int tid = threadIdx.x, wave = tid >> 5, l = tid & 31;
    int lrow = l & 15, hi = (l >= 16) ? 1 : 0;
    int m0 = wave * 16;

    __builtin_prefetch(wof, 0, 2);

    const _Float16* ab = at + (size_t)w * 64 * 128;
    for (int it = 0; it < 8; ++it) {        // 16KB tile: 1024 b128
        int idx = tid + it * 128;
        async_ld_b128(&((uint4*)As)[idx], (const uint4*)ab + idx);
    }
    wait_async0();
    __syncthreads();

    Frag a[4];
    for (int kk = 0; kk < 4; ++kk)
        for (int r = 0; r < 8; ++r) {
            int kb = kk * 32 + 2 * r + (hi ? 16 : 0);
            a[kk].u[r] = *(const unsigned*)&As[m0 + lrow][kb];
        }

    for (int t = 0; t < 8; ++t) {
        v8f acc = {};
        for (int kk = 0; kk < 4; ++kk) {
            Frag b;
            const uint4* bp = (const uint4*)(wof + ((size_t)(t * 4 + kk) * 32 + l) * 8);
            *(uint4*)&b.u[0] = bp[0];
            *(uint4*)&b.u[4] = bp[1];
            acc = wmma_f16(a[kk].v, b.v, acc);
        }
        int n = t * 16 + lrow;
        float bb = bout[n];
        for (int r = 0; r < 8; ++r) {
            int m = m0 + r + (hi ? 8 : 0);
            int iz = m >> 4, iy = (m >> 2) & 3, ix = m & 3;
            int z = (wh * 4 + iz + 2) & 63;
            int y = (ww * 4 + iy + 2) & 63;
            int xd = (wd * 4 + ix + 2) & 63;
            out[(((size_t)z * 64 + y) * 64 + xd) * 128 + n] = acc[r] + bb;
        }
    }
}

extern "C" void kernel_launch(void* const* d_in, const int* in_sizes, int n_in,
                              void* d_out, int out_size, void* d_ws, size_t ws_size,
                              hipStream_t stream) {
    const float* x    = (const float*)d_in[0];
    const float* Wqkv = (const float*)d_in[1];
    const float* pos  = (const float*)d_in[2];
    const float* Wout = (const float*)d_in[3];
    const float* bout = (const float*)d_in[4];
    float* out = (float*)d_out;
    char* ws = (char*)d_ws;

    _Float16* q   = (_Float16*)(ws + OFF_Q);
    _Float16* k   = (_Float16*)(ws + OFF_K);
    _Float16* v   = (_Float16*)(ws + OFF_V);
    _Float16* at  = (_Float16*)(ws + OFF_AT);
    unsigned* wqf = (unsigned*)(ws + OFF_WQ);
    unsigned* wof = (unsigned*)(ws + OFF_WO);
    float*    bia = (float*)(ws + OFF_B);

    prep_kernel<<<144, 256, 0, stream>>>(Wqkv, Wout, pos, wqf, wof, bia);
    qkv_kernel<<<4096, 256, 0, stream>>>(x, wqf, q, k, v);
    attn_kernel<<<16384, 128, 0, stream>>>(q, k, v, bia, at);
    proj_kernel<<<4096, 128, 0, stream>>>(at, wof, bout, out);
}
